// MOST_scan_8881992368858
// MI455X (gfx1250) — matrix operand order repaired and checked by reading, not compiled, not run on hardware
//
#include <hip/hip_runtime.h>

#define DIM 20
#define CH 300
#define DEPTH 6
#define HID 80
#define SLICE 20
#define FRAME 15
#define BATCH 4096
#define NB 4
#define NCOLS (NB * DIM)   // 80  : N of token-mix (4 batch x 20 feat)
#define KTOK 320           // padded token dimension (M and K of token-mix)
#define KD 32              // padded DIM as K (MLP1)
#define KH 96              // padded HID as K (MLP2)
#define ND 32              // padded DIM as N (MLP2)
#define MMLP (NB * CH)     // 1200 : M of MLP ((batch,chan) pairs)

#define TOKW_ELEMS (KTOK * KTOK)       // 102400 per (branch, depth)
#define M1T_ELEMS  (HID * KD)          // 2560   [n=hid][k=dim_pad]
#define M2T_ELEMS  (ND * KH)           // 3072   [n=dim_pad][k=hid_pad]
#define TOKW_TOTAL (3 * DEPTH * TOKW_ELEMS)
#define M1T_TOTAL  (3 * DEPTH * M1T_ELEMS)
#define M2T_TOTAL  (3 * DEPTH * M2T_ELEMS)

typedef _Float16 v8h  __attribute__((ext_vector_type(8)));
typedef _Float16 v16h __attribute__((ext_vector_type(16)));
typedef float    v8f  __attribute__((ext_vector_type(8)));

__device__ __forceinline__ v16h cat8(v8h lo, v8h hi) {
  return __builtin_shufflevector(lo, hi, 0, 1, 2, 3, 4, 5, 6, 7,
                                 8, 9, 10, 11, 12, 13, 14, 15);
}

__device__ __forceinline__ v8f wmma_f16(v16h a, v16h b, v8f c) {
  return __builtin_amdgcn_wmma_f32_16x16x32_f16(false, a, false, b,
                                                (short)0, c, false, false);
}

// ---------------------------------------------------------------------------
// Kernel 1: convert fp32 weights -> padded / (for B-operands) transposed f16
// ---------------------------------------------------------------------------
struct PrepArgs {
  const float* tok_w[3];   // [DEPTH, CH, CH]
  const float* m1_w[3];    // [DEPTH, DIM, HID]
  const float* m2_w[3];    // [DEPTH, HID, DIM]
  _Float16* tokW;          // [3, DEPTH, KTOK, KTOK]  row-major (A operand)
  _Float16* m1T;           // [3, DEPTH, HID, KD]     (B operand, [N][K])
  _Float16* m2T;           // [3, DEPTH, ND, KH]      (B operand, [N][K])
};

__global__ void prep_weights(PrepArgs P) {
  const int total = TOKW_TOTAL + M1T_TOTAL + M2T_TOTAL;
  for (int i = blockIdx.x * blockDim.x + threadIdx.x; i < total;
       i += gridDim.x * blockDim.x) {
    if (i < TOKW_TOTAL) {
      int bd = i / TOKW_ELEMS, r = i % TOKW_ELEMS;
      int br = bd / DEPTH, dep = bd % DEPTH;
      int e = r / KTOK, c = r % KTOK;
      float v = (e < CH && c < CH) ? P.tok_w[br][(dep * CH + e) * CH + c] : 0.f;
      P.tokW[i] = (_Float16)v;
    } else if (i < TOKW_TOTAL + M1T_TOTAL) {
      int j = i - TOKW_TOTAL;
      int bd = j / M1T_ELEMS, r = j % M1T_ELEMS;
      int br = bd / DEPTH, dep = bd % DEPTH;
      int n = r / KD, k = r % KD;   // n: hid, k: dim
      float v = (k < DIM) ? P.m1_w[br][(dep * DIM + k) * HID + n] : 0.f;
      P.m1T[j] = (_Float16)v;
    } else {
      int j = i - TOKW_TOTAL - M1T_TOTAL;
      int bd = j / M2T_ELEMS, r = j % M2T_ELEMS;
      int br = bd / DEPTH, dep = bd % DEPTH;
      int n = r / KH, k = r % KH;   // n: dim, k: hid
      float v = (n < DIM && k < HID) ? P.m2_w[br][(dep * HID + k) * DIM + n] : 0.f;
      P.m2T[j] = (_Float16)v;
    }
  }
}

// ---------------------------------------------------------------------------
// Kernel 2: fused 6-deep ResMLP branch, NB=4 batch elements per workgroup.
// LDS: H fp32 [CH][NCOLS] residual; U f16 union (token-mix B / MLP A);
//      SW f16 per-wave 16x96 transpose bounce.
// ---------------------------------------------------------------------------
struct MainArgs {
  const float* x;
  const float* a1_g[3]; const float* a1_b[3];
  const float* tok_b[3];
  const float* a2_g[3]; const float* a2_b[3];
  const float* m1_b[3]; const float* m2_b[3];
  const float* af_g[3]; const float* af_b[3];
  const _Float16* tokW; const _Float16* m1T; const _Float16* m2T;
  float* out;
};

#define LDS_H_BYTES  (CH * NCOLS * 4)        // 96000
#define LDS_U_BYTES  (MMLP * KD * 2)         // 76800 (>= NCOLS*KTOK*2, union)
#define LDS_SW_BYTES (8 * 16 * KH * 2)       // 24576
#define LDS_TOTAL    (LDS_H_BYTES + LDS_U_BYTES + LDS_SW_BYTES)

__launch_bounds__(256, 1)
__global__ void resmlp_fused(MainArgs A) {
  extern __shared__ char smem_raw[];
  float*    H  = (float*)smem_raw;                               // [CH][NCOLS]
  _Float16* U  = (_Float16*)(smem_raw + LDS_H_BYTES);            // union
  _Float16* SW = (_Float16*)(smem_raw + LDS_H_BYTES + LDS_U_BYTES);

  const int tid = threadIdx.x;
  const int wave = tid >> 5;
  const int lane = tid & 31;
  const int br = blockIdx.y;
  const int b0 = blockIdx.x * NB;

  _Float16* mySW = SW + wave * 16 * KH;
  // zero the K-pad columns (hid 80..95) of this wave's transpose buffer once
  for (int i = lane; i < 16 * 16; i += 32) {
    mySW[(i / 16) * KH + HID + (i % 16)] = (_Float16)0.f;
  }

  // ---- load x into residual H (branch token permutation) via async DMA ----
  // H lives at the start of dynamic LDS; its LDS byte address is
  // groupstaticsize() + i*4.  Per-lane global gather -> per-lane LDS write,
  // tracked by ASYNCcnt.
  {
    unsigned ldsbase = __builtin_amdgcn_groupstaticsize();
    for (int i = tid; i < CH * NCOLS; i += 256) {
      int c = i / NCOLS, n = i % NCOLS;
      int bq = n / DIM, dd = n % DIM;
      int s, t;
      if (br == 0)      { s = c / FRAME;              t = c % FRAME; }
      else if (br == 1) { s = SLICE - 1 - c / FRAME;  t = FRAME - 1 - c % FRAME; }
      else              { s = c % SLICE;              t = c / SLICE; }
      const float* src = A.x + (((b0 + bq) * SLICE + s) * FRAME + t) * DIM + dd;
      unsigned dst = ldsbase + (unsigned)i * 4u;
      asm volatile("global_load_async_to_lds_b32 %0, %1, off"
                   :: "v"(dst), "v"(src) : "memory");
    }
    asm volatile("s_wait_asynccnt 0x0" ::: "memory");
  }
  __syncthreads();

  const _Float16* tokWbr = A.tokW + (br * DEPTH) * TOKW_ELEMS;
  const int arow = (lane & 15);
  const int aoff = (lane >> 4) * 16;
  const int boff = (lane >> 4) * 16;

  // hoisted loop-invariant MLP B-operand fragments
  v16h W1f[5];
  v16h W2f[3][2];

  for (int dep = 0; dep < DEPTH; ++dep) {
    // ---- Phase A: affine1 -> f16 token-mix B-operand, stored [N][K] ----
    const float* g1 = A.a1_g[br] + dep * DIM;
    const float* c1 = A.a1_b[br] + dep * DIM;
    for (int i = tid; i < NCOLS * KTOK; i += 256) {
      int n = i / KTOK, k = i % KTOK;
      float v = 0.f;
      if (k < CH) { int dd = n % DIM; v = g1[dd] * H[k * NCOLS + n] + c1[dd]; }
      U[i] = (_Float16)v;
    }
    __syncthreads();

    // ---- Phase B: token-mix matmul, n-register-blocked: one A fragment
    //      feeds all 5 N-tiles.  Gather the 5 B fragments first so the 5
    //      WMMAs issue back-to-back behind a single dscnt wait. ----
    const _Float16* W = tokWbr + dep * TOKW_ELEMS;
    const float* tb = A.tok_b[br] + dep * CH;
    for (int mt = wave; mt < 19; mt += 8) {
      v8f acc[5] = {};
      const _Float16* pw = W + (mt * 16 + arow) * KTOK + aoff;
      const _Float16* pu = U + arow * KTOK + boff;
#pragma unroll
      for (int k = 0; k < KTOK / 32; ++k) {
        const _Float16* pa = pw + k * 32;
        __builtin_prefetch(pa + 32, 0, 0);
        v16h Bf[5];
#pragma unroll
        for (int nt = 0; nt < 5; ++nt)
          Bf[nt] = *(const v16h*)(pu + nt * (16 * KTOK) + k * 32);
        v16h Af = cat8(*(const v8h*)pa, *(const v8h*)(pa + 8));
#pragma unroll
        for (int nt = 0; nt < 5; ++nt)
          acc[nt] = wmma_f16(Af, Bf[nt], acc[nt]);
      }
      int m0 = mt * 16 + (lane >> 4) * 8;
#pragma unroll
      for (int nt = 0; nt < 5; ++nt) {
        int nc = nt * 16 + (lane & 15);
#pragma unroll
        for (int r = 0; r < 8; ++r) {
          int e = m0 + r;
          if (e < CH) H[e * NCOLS + nc] += acc[nt][r] + tb[e];
        }
      }
    }
    __syncthreads();

    // ---- Phase C: affine2 -> f16 MLP A-operand Z, stored [M][Kpad32] ----
    const float* g2 = A.a2_g[br] + dep * DIM;
    const float* c2 = A.a2_b[br] + dep * DIM;
    for (int i = tid; i < MMLP * KD; i += 256) {
      int m = i / KD, k = i % KD;
      float v = 0.f;
      if (k < DIM) {
        int bq = m / CH, c = m % CH;
        v = g2[k] * H[c * NCOLS + bq * DIM + k] + c2[k];
      }
      U[i] = (_Float16)v;
    }
    __syncthreads();

    // ---- Phase D: MLP  relu(Z@m1+b1)@m2+b2, fused per 16-row M-tile ----
    const _Float16* W1 = A.m1T + (br * DEPTH + dep) * M1T_ELEMS;
    const _Float16* W2 = A.m2T + (br * DEPTH + dep) * M2T_ELEMS;
    const float* mb1 = A.m1_b[br] + dep * HID;
    const float* mb2 = A.m2_b[br] + dep * DIM;
    // B-fragments are invariant across the 75 M-tiles: load once
#pragma unroll
    for (int nt = 0; nt < 5; ++nt)
      W1f[nt] = *(const v16h*)(W1 + (nt * 16 + (lane & 15)) * KD + boff);
#pragma unroll
    for (int k = 0; k < KH / 32; ++k) {
      W2f[k][0] = *(const v16h*)(W2 + (lane & 15) * KH + k * 32 + boff);
      W2f[k][1] = *(const v16h*)(W2 + (16 + (lane & 15)) * KH + k * 32 + boff);
    }
    for (int mt = wave; mt < MMLP / 16; mt += 8) {
      int m0 = mt * 16;
      // first GEMM: Z-tile (K=32 single step) x 5 hid N-tiles, relu,
      // transpose-bounce f16 into mySW[16][96]
      {
        const _Float16* pz = U + (m0 + arow) * KD + aoff;
        v16h Zf = cat8(*(const v8h*)pz, *(const v8h*)(pz + 8));
        v8f hacc[5];
#pragma unroll
        for (int nt = 0; nt < 5; ++nt) {
          v8f zero = {};
          hacc[nt] = wmma_f16(Zf, W1f[nt], zero);
        }
        int mloc0 = (lane >> 4) * 8;
#pragma unroll
        for (int nt = 0; nt < 5; ++nt) {
          int n = nt * 16 + (lane & 15);
          float bias = mb1[n];
#pragma unroll
          for (int r = 0; r < 8; ++r) {
            float hv = hacc[nt][r] + bias;
            hv = hv > 0.f ? hv : 0.f;
            mySW[(mloc0 + r) * KH + n] = (_Float16)hv;
          }
        }
      }
      asm volatile("s_wait_dscnt 0x0" ::: "memory"); // cross-lane LDS handoff
      // second GEMM: hid-tile (K=96, 3 steps) x 2 dim N-tiles
      v8f acc0 = {}, acc1 = {};
#pragma unroll
      for (int k = 0; k < KH / 32; ++k) {
        const _Float16* ps = mySW + arow * KH + k * 32 + aoff;
        v16h Af = cat8(*(const v8h*)ps, *(const v8h*)(ps + 8));
        acc0 = wmma_f16(Af, W2f[k][0], acc0);
        acc1 = wmma_f16(Af, W2f[k][1], acc1);
      }
      int mloc0 = (lane >> 4) * 8;
      int dd0 = lane & 15;
#pragma unroll
      for (int r = 0; r < 8; ++r) {
        int m = m0 + mloc0 + r;
        int bq = m / CH, c = m % CH;
        H[c * NCOLS + bq * DIM + dd0] += acc0[r] + mb2[dd0];
        int dd1 = 16 + dd0;
        if (dd1 < DIM) H[c * NCOLS + bq * DIM + dd1] += acc1[r] + mb2[dd1];
      }
    }
    __syncthreads();
  }

  // ---- final affine + mean over DIM ----
  const float* afg = A.af_g[br];
  const float* afb = A.af_b[br];
  for (int i = tid; i < MMLP; i += 256) {
    int bq = i / CH, c = i % CH;
    float sum = 0.f;
#pragma unroll
    for (int dd = 0; dd < DIM; ++dd)
      sum += afg[dd] * H[c * NCOLS + bq * DIM + dd] + afb[dd];
    A.out[(b0 + bq) * (3 * CH) + br * CH + c] = sum * (1.0f / DIM);
  }
}

// ---------------------------------------------------------------------------
extern "C" void kernel_launch(void* const* d_in, const int* in_sizes, int n_in,
                              void* d_out, int out_size, void* d_ws, size_t ws_size,
                              hipStream_t stream) {
  (void)in_sizes; (void)n_in; (void)out_size; (void)ws_size;

  _Float16* ws16 = (_Float16*)d_ws;
  _Float16* tokW = ws16;
  _Float16* m1T  = ws16 + TOKW_TOTAL;
  _Float16* m2T  = m1T + M1T_TOTAL;

  PrepArgs pa;
  MainArgs ma;
  ma.x = (const float*)d_in[0];
  for (int br = 0; br < 3; ++br) {
    void* const* p = d_in + 1 + br * 12;  // a1_g,a1_b,tok_w,tok_b,a2_g,a2_b,
                                          // m1_w,m1_b,m2_w,m2_b,af_g,af_b
    ma.a1_g[br]  = (const float*)p[0];
    ma.a1_b[br]  = (const float*)p[1];
    pa.tok_w[br] = (const float*)p[2];
    ma.tok_b[br] = (const float*)p[3];
    ma.a2_g[br]  = (const float*)p[4];
    ma.a2_b[br]  = (const float*)p[5];
    pa.m1_w[br]  = (const float*)p[6];
    ma.m1_b[br]  = (const float*)p[7];
    pa.m2_w[br]  = (const float*)p[8];
    ma.m2_b[br]  = (const float*)p[9];
    ma.af_g[br]  = (const float*)p[10];
    ma.af_b[br]  = (const float*)p[11];
  }
  pa.tokW = tokW; pa.m1T = m1T; pa.m2T = m2T;
  ma.tokW = tokW; ma.m1T = m1T; ma.m2T = m2T;
  ma.out = (float*)d_out;

  const int total = TOKW_TOTAL + M1T_TOTAL + M2T_TOTAL;
  prep_weights<<<(total + 255) / 256, 256, 0, stream>>>(pa);
  resmlp_fused<<<dim3(BATCH / NB, 3), 256, LDS_TOTAL, stream>>>(ma);
}
